// GlobalLocalAttentionHead_58299886076329
// MI455X (gfx1250) — matrix operand order repaired and checked by reading, not compiled, not run on hardware
//
#include <hip/hip_runtime.h>
#include <hip/hip_bf16.h>
#include <math.h>

// ---------------------------------------------------------------------------
// Problem constants (reference: B=4, S=2048, D=64, t=1.0, softmax over axis=1)
// ---------------------------------------------------------------------------
#define BB 4
#define SS 2048
#define DD 64
#define JSPLIT 8
#define RSQRT_S 0.022097086912079608f   // 1/sqrt(2048)

typedef __attribute__((ext_vector_type(16))) _Float16 v16h;
typedef __attribute__((ext_vector_type(8)))  float    v8f;

// ---------------------------------------------------------------------------
// WMMA fragment loaders (CDNA5 ISA 7.12.2 layouts, 16-bit A 16x32 / B 32x16).
// A fragment: lane l -> row m0+(l&15); halves 0..7 at K=k0+8*hi, 8..15 at +16.
// B fragment: lane l -> col n0+(l&15); halves 0..7 at K=k0+8*hi, 8..15 at +16
// (B is stored as Brow[N][K], i.e. B^T row-major).
// ---------------------------------------------------------------------------
static __device__ __forceinline__ v16h load_frag_a(const _Float16* __restrict__ A,
                                                   int ldA, int m0, int k0, int lane) {
  const int m  = m0 + (lane & 15);
  const int hi = lane >> 4;
  const _Float16* p = A + (size_t)m * ldA + k0 + 8 * hi;
  union { v16h h; float4 f[2]; } u;
  u.f[0] = *(const float4*)(p);
  u.f[1] = *(const float4*)(p + 16);
  return u.h;
}

static __device__ __forceinline__ v16h load_frag_b(const _Float16* __restrict__ Brow,
                                                   int ldB, int n0, int k0, int lane) {
  const int n  = n0 + (lane & 15);
  const int hi = lane >> 4;
  const _Float16* p = Brow + (size_t)n * ldB + k0 + 8 * hi;
  union { v16h h; float4 f[2]; } u;
  u.f[0] = *(const float4*)(p);
  u.f[1] = *(const float4*)(p + 16);
  return u.h;
}

static __device__ __forceinline__ v8f wmma_f16(v16h a, v16h b, v8f c) {
  return __builtin_amdgcn_wmma_f32_16x16x32_f16(false, a, false, b, (short)0, c,
                                                false, false);
}

// ---------------------------------------------------------------------------
// Generic WMMA GEMM:  C = A[M,K] * Brow[N,K]^T  (f16 in, f32 acc)
// One wave computes a 16x64 strip; 4 waves per block cover 64x64.
// MODE 0: store f16 C[M,N]        (Q)
// MODE 1: store f16 C^T as [N,M]  (V^T, so later B-fragments are contiguous)
// ---------------------------------------------------------------------------
template <int MODE>
__global__ __launch_bounds__(128) void gemm_abt_kernel(
    const _Float16* __restrict__ Abase, const _Float16* __restrict__ Bbase,
    _Float16* __restrict__ Cbase, int M, int N, int K,
    long aStride, long bStride, long cStride) {
  const int lane = threadIdx.x & 31;          // wave32
  const int wave = threadIdx.x >> 5;
  const int b    = blockIdx.z;
  const int m0   = blockIdx.x * 64 + wave * 16;
  const int n0   = blockIdx.y * 64;

  const _Float16* A  = Abase + (size_t)b * aStride;
  const _Float16* Br = Bbase + (size_t)b * bStride;

  v8f acc[4] = {};
  for (int k0 = 0; k0 < K; k0 += 32) {
    if (k0 + 32 < K)  // warm caches for the next A slab (global_prefetch_b8)
      __builtin_prefetch(A + (size_t)(m0 + (lane & 15)) * K + k0 + 32 + 8 * (lane >> 4), 0, 0);
    v16h a = load_frag_a(A, K, m0, k0, lane);
#pragma unroll
    for (int t = 0; t < 4; ++t)
      acc[t] = wmma_f16(a, load_frag_b(Br, K, n0 + 16 * t, k0, lane), acc[t]);
  }

  const int nl = lane & 15, hi = lane >> 4;
#pragma unroll
  for (int t = 0; t < 4; ++t) {
    const int n = n0 + 16 * t + nl;
#pragma unroll
    for (int r = 0; r < 8; ++r) {
      const int m = m0 + 8 * hi + r;
      const float v = acc[t][r];
      if (MODE == 0)
        Cbase[(size_t)b * cStride + (size_t)m * N + n] = (_Float16)v;
      else
        Cbase[(size_t)b * cStride + (size_t)n * M + m] = (_Float16)v;
    }
  }
}

// ---------------------------------------------------------------------------
// Prep: per (b,d) column, block-scan cumsum over S to get G = cumsum/(pos+1);
// emit f16 copies: xh [B,S,D], xT [B,D,S], GT [B,D,S].
// ---------------------------------------------------------------------------
__global__ __launch_bounds__(256) void prep_kernel(const float* __restrict__ x,
                                                   _Float16* __restrict__ xh,
                                                   _Float16* __restrict__ xT,
                                                   _Float16* __restrict__ GT) {
  const int b = blockIdx.x / DD;
  const int d = blockIdx.x % DD;
  const int t = threadIdx.x;

  float xv[8], pre[8];
  float run = 0.0f;
#pragma unroll
  for (int e = 0; e < 8; ++e) {
    const int s = t * 8 + e;
    xv[e] = x[((size_t)b * SS + s) * DD + d];
    run += xv[e];
    pre[e] = run;
  }

  __shared__ float ssum[256];
  ssum[t] = run;
  __syncthreads();
  for (int off = 1; off < 256; off <<= 1) {   // Hillis-Steele inclusive scan
    float add = (t >= off) ? ssum[t - off] : 0.0f;
    __syncthreads();
    ssum[t] += add;
    __syncthreads();
  }
  const float base = ssum[t] - run;           // exclusive prefix

#pragma unroll
  for (int e = 0; e < 8; ++e) {
    const int s = t * 8 + e;
    const float g = (base + pre[e]) / (float)(s + 1);
    xh[((size_t)b * SS + s) * DD + d] = (_Float16)xv[e];
    xT[((size_t)b * DD + d) * SS + s] = (_Float16)xv[e];
    GT[((size_t)b * DD + d) * SS + s] = (_Float16)g;
  }
}

// f32 -> f16 conversion of both weight matrices
__global__ __launch_bounds__(256) void wconv_kernel(const float* __restrict__ Wq,
                                                    const float* __restrict__ Wv,
                                                    _Float16* __restrict__ Wqh,
                                                    _Float16* __restrict__ Wvh) {
  const size_t i = (size_t)blockIdx.x * blockDim.x + threadIdx.x;
  Wqh[i] = (_Float16)Wq[i];
  Wvh[i] = (_Float16)Wv[i];
}

// ---------------------------------------------------------------------------
// Score recompute passes (the SxS score matrix is never written to memory).
// Each wave recomputes a 16x64 masked score strip via WMMA, then reduces it
// per column j: PASS 0 -> max, PASS 1 -> sum of exp((v - colmax)/sqrt(S)).
// Per-block partials land in part[gridDim.x][B*S]; a deterministic tree
// reduction finishes the column statistic (no float atomics -> replay-stable).
// ---------------------------------------------------------------------------
template <int PASS>
__global__ __launch_bounds__(128) void score_pass_kernel(
    const _Float16* __restrict__ Qh, const _Float16* __restrict__ xh,
    const float* __restrict__ cmax, float* __restrict__ part) {
  const int lane = threadIdx.x & 31, wave = threadIdx.x >> 5;
  const int b  = blockIdx.z;
  const int i0 = blockIdx.x * 64 + wave * 16;
  const int j0 = blockIdx.y * 64;
  const _Float16* Q = Qh + (size_t)b * SS * DD;
  const _Float16* X = xh + (size_t)b * SS * DD;

  v8f sc[4] = {};
#pragma unroll
  for (int k0 = 0; k0 < DD; k0 += 32) {
    v16h a = load_frag_a(Q, DD, i0, k0, lane);
#pragma unroll
    for (int t = 0; t < 4; ++t)
      sc[t] = wmma_f16(a, load_frag_b(X, DD, j0 + 16 * t, k0, lane), sc[t]);
  }

  const int nl = lane & 15, hi = lane >> 4;
  __shared__ float smem[4][64];
#pragma unroll
  for (int t = 0; t < 4; ++t) {
    const int j = j0 + 16 * t + nl;
    const float cm = (PASS == 1) ? cmax[(size_t)b * SS + j] : 0.0f;
    float red = (PASS == 0) ? -INFINITY : 0.0f;
#pragma unroll
    for (int r = 0; r < 8; ++r) {
      const int i = i0 + 8 * hi + r;
      const float v = sc[t][r];
      const bool live = (v > 1.0f) && (j <= i);   // sigma threshold + causal
      if (PASS == 0) {
        if (live) red = fmaxf(red, v);
      } else {
        red += live ? __expf((v - cm) * RSQRT_S) : 0.0f;
      }
    }
    const float o = __shfl_xor(red, 16, 32);      // merge the two lane halves
    red = (PASS == 0) ? fmaxf(red, o) : red + o;
    if (lane < 16) smem[wave][t * 16 + nl] = red;
  }
  __syncthreads();
  if (threadIdx.x < 64) {
    const float a0 = smem[0][threadIdx.x], a1 = smem[1][threadIdx.x];
    const float a2 = smem[2][threadIdx.x], a3 = smem[3][threadIdx.x];
    const float r = (PASS == 0) ? fmaxf(fmaxf(a0, a1), fmaxf(a2, a3))
                                : (a0 + a1) + (a2 + a3);
    part[(size_t)blockIdx.x * (BB * SS) + (size_t)b * SS + j0 + threadIdx.x] = r;
  }
}

// Deterministic finish of the column statistic over the 32 i-blocks.
template <int OP>
__global__ __launch_bounds__(256) void reduce_part_kernel(
    const float* __restrict__ part, float* __restrict__ outv) {
  const size_t c = (size_t)blockIdx.x * 256 + threadIdx.x;  // < B*S
  float r = (OP == 0) ? -INFINITY : 0.0f;
#pragma unroll
  for (int p = 0; p < SS / 64; ++p) {
    const float v = part[(size_t)p * (BB * SS) + c];
    r = (OP == 0) ? fmaxf(r, v) : r + v;
  }
  outv[c] = r;
}

// ---------------------------------------------------------------------------
// Fused output: out = softmax_col(masked QK^T) @ V without materializing P.
// Per wave: one 16-row query tile; loop j in 32-steps: WMMA score tile ->
// masked exp * rcp(colsum[j]) -> LDS transpose (C-layout -> A-fragment) ->
// WMMA against V^T. 8 j-chunks write partial outputs (reduced afterwards).
// ---------------------------------------------------------------------------
__global__ __launch_bounds__(128) void attn_out_kernel(
    const _Float16* __restrict__ Qh, const _Float16* __restrict__ xh,
    const _Float16* __restrict__ Vt, const float* __restrict__ cmax,
    const float* __restrict__ csum, float* __restrict__ Opart) {
  const int lane = threadIdx.x & 31, wave = threadIdx.x >> 5;
  const int b  = blockIdx.z;
  const int i0 = blockIdx.x * 64 + wave * 16;
  const int jc = blockIdx.y;                       // j-chunk 0..JSPLIT-1
  const _Float16* Q = Qh + (size_t)b * SS * DD;
  const _Float16* X = xh + (size_t)b * SS * DD;
  const _Float16* V = Vt + (size_t)b * DD * SS;
  const int nl = lane & 15, hi = lane >> 4;

  v16h aq[2];
  aq[0] = load_frag_a(Q, DD, i0, 0, lane);
  aq[1] = load_frag_a(Q, DD, i0, 32, lane);

  // padded 16x32 f16 tile per wave (96B row stride: 16B-aligned b128 LDS loads)
  __shared__ __align__(16) _Float16 et[4][16][48];

  v8f acc[4] = {};
  for (int jt = 0; jt < SS / (32 * JSPLIT); ++jt) {   // 8 iterations
    const int j0 = jc * (SS / JSPLIT) + jt * 32;

    // --- score tile 16 x 32 (two 16-wide column tiles) ---
    v8f sc[2] = {};
#pragma unroll
    for (int k = 0; k < 2; ++k) {
#pragma unroll
      for (int u = 0; u < 2; ++u)
        sc[u] = wmma_f16(aq[k], load_frag_b(X, DD, j0 + 16 * u, 32 * k, lane), sc[u]);
    }

    // --- masked, normalized exp -> LDS (row = local i, col = local j) ---
#pragma unroll
    for (int u = 0; u < 2; ++u) {
      const int j = j0 + 16 * u + nl;
      const float cm  = cmax[(size_t)b * SS + j];
      // fast v_rcp_f32: csum is a well-conditioned positive sum, and the
      // result is rounded to f16 anyway — avoids the IEEE div sequence
      // stalling the WMMA pipe.
      const float rcs = __builtin_amdgcn_rcpf(csum[(size_t)b * SS + j]);
#pragma unroll
      for (int r = 0; r < 8; ++r) {
        const int i = i0 + 8 * hi + r;
        const float v = sc[u][r];
        const float e = ((v > 1.0f) && (j <= i))
                            ? __expf((v - cm) * RSQRT_S) * rcs
                            : 0.0f;
        et[wave][8 * hi + r][16 * u + nl] = (_Float16)e;
      }
    }
    __syncthreads();

    // --- reload as an A-fragment (transpose through LDS) ---
    union { v16h h; float4 f[2]; } ua;
    ua.f[0] = *(const float4*)&et[wave][nl][8 * hi];
    ua.f[1] = *(const float4*)&et[wave][nl][8 * hi + 16];
    __syncthreads();

    // --- accumulate E-tile @ V^T ---
#pragma unroll
    for (int t = 0; t < 4; ++t)
      acc[t] = wmma_f16(ua.h, load_frag_b(V, SS, 16 * t, j0, lane), acc[t]);
  }

  float* Op = Opart + (size_t)jc * (BB * SS * DD) + (size_t)b * SS * DD;
#pragma unroll
  for (int t = 0; t < 4; ++t) {
    const int n = 16 * t + nl;
#pragma unroll
    for (int r = 0; r < 8; ++r)
      Op[(size_t)(i0 + 8 * hi + r) * DD + n] = acc[t][r];
  }
}

// Deterministic sum of the JSPLIT partial outputs.
__global__ __launch_bounds__(256) void reduce_out_kernel(
    const float* __restrict__ Opart, float* __restrict__ out) {
  const size_t idx = (size_t)blockIdx.x * 256 + threadIdx.x;  // < B*S*D
  float s = 0.0f;
#pragma unroll
  for (int p = 0; p < JSPLIT; ++p)
    s += Opart[(size_t)p * (BB * SS * DD) + idx];
  out[idx] = s;
}

// ---------------------------------------------------------------------------
// Launcher
// ---------------------------------------------------------------------------
extern "C" void kernel_launch(void* const* d_in, const int* in_sizes, int n_in,
                              void* d_out, int out_size, void* d_ws, size_t ws_size,
                              hipStream_t stream) {
  const float* x  = (const float*)d_in[0];   // [B,S,D]
  const float* Wq = (const float*)d_in[1];   // [S,S]
  const float* Wv = (const float*)d_in[2];   // [S,S]
  float* out = (float*)d_out;                // [B,S,D]

  char* w = (char*)d_ws;
  auto alloc = [&](size_t bytes) -> void* {
    void* p = (void*)w;
    w += (bytes + 255) & ~(size_t)255;
    return p;
  };
  _Float16* Wqh  = (_Float16*)alloc((size_t)SS * SS * 2);            // 8 MB
  _Float16* Wvh  = (_Float16*)alloc((size_t)SS * SS * 2);            // 8 MB
  _Float16* xh   = (_Float16*)alloc((size_t)BB * SS * DD * 2);       // 1 MB [B,S,D]
  _Float16* xT   = (_Float16*)alloc((size_t)BB * DD * SS * 2);       // 1 MB [B,D,S]
  _Float16* GT   = (_Float16*)alloc((size_t)BB * DD * SS * 2);       // 1 MB [B,D,S]
  _Float16* Qh   = (_Float16*)alloc((size_t)BB * SS * DD * 2);       // 1 MB [B,S,D]
  _Float16* Vt   = (_Float16*)alloc((size_t)BB * DD * SS * 2);       // 1 MB [B,D,S]
  float*    part = (float*)   alloc((size_t)(SS / 64) * BB * SS * 4);// 1 MB (reused)
  float*    cmx  = (float*)   alloc((size_t)BB * SS * 4);
  float*    csm  = (float*)   alloc((size_t)BB * SS * 4);
  float*    Opar = (float*)   alloc((size_t)JSPLIT * BB * SS * DD * 4); // 16 MB

  // 1) G (cumsum/pos) + f16 copies of x / x^T / G^T
  prep_kernel<<<BB * DD, 256, 0, stream>>>(x, xh, xT, GT);
  // 2) weights -> f16
  wconv_kernel<<<(SS * SS) / 256, 256, 0, stream>>>(Wq, Wv, Wqh, Wvh);
  // 3) V^T = (W_V @ x)^T  [B,D,S]
  gemm_abt_kernel<1><<<dim3(SS / 64, 1, BB), 128, 0, stream>>>(
      Wvh, xT, Vt, SS, DD, SS, 0L, (long)DD * SS, (long)DD * SS);
  // 4) Q = W_Q @ G        [B,S,D]
  gemm_abt_kernel<0><<<dim3(SS / 64, 1, BB), 128, 0, stream>>>(
      Wqh, GT, Qh, SS, DD, SS, 0L, (long)DD * SS, (long)SS * DD);
  // 5) column softmax statistics by score recompute (no SxS materialization)
  score_pass_kernel<0><<<dim3(SS / 64, SS / 64, BB), 128, 0, stream>>>(Qh, xh, cmx, part);
  reduce_part_kernel<0><<<(BB * SS) / 256, 256, 0, stream>>>(part, cmx);
  score_pass_kernel<1><<<dim3(SS / 64, SS / 64, BB), 128, 0, stream>>>(Qh, xh, cmx, part);
  reduce_part_kernel<1><<<(BB * SS) / 256, 256, 0, stream>>>(part, csm);
  // 6) fused P@V with on-the-fly P, partials over JSPLIT j-chunks
  attn_out_kernel<<<dim3(SS / 64, JSPLIT, BB), 128, 0, stream>>>(Qh, xh, Vt, cmx, csm, Opar);
  reduce_out_kernel<<<(BB * SS * DD) / 256, 256, 0, stream>>>(Opar, out);
}
